// EnhancedGNN_6433861009953
// MI455X (gfx1250) — compile-verified, matrix operand order checked by reading, not run on hardware
//
#include <hip/hip_runtime.h>
#include <cmath>

typedef __attribute__((ext_vector_type(2)))  float    v2f;
typedef __attribute__((ext_vector_type(8)))  float    v8f;
typedef __attribute__((ext_vector_type(16))) _Float16 v16h;

#ifndef __has_builtin
#define __has_builtin(x) 0
#endif

#if __has_builtin(__builtin_amdgcn_wmma_f32_16x16x4_f32)
#define GCN_USE_F32_WMMA 1
#else
#define GCN_USE_F32_WMMA 0
#endif

// ---------------------------------------------------------------------------
// Degree / normalization kernels
// ---------------------------------------------------------------------------
__global__ __launch_bounds__(256) void k_init_deg(float* __restrict__ deg, int N) {
    int i = blockIdx.x * blockDim.x + threadIdx.x;
    if (i < N) deg[i] = 1.0f;              // self-loop contributes 1 to every degree
}

__global__ __launch_bounds__(256) void k_deg_accum(const int* __restrict__ dst,
                                                   float* __restrict__ deg, int E) {
    int e = blockIdx.x * blockDim.x + threadIdx.x;
    if (e < E) atomicAdd(&deg[dst[e]], 1.0f);
}

__global__ __launch_bounds__(256) void k_rsqrt_inplace(float* __restrict__ deg, int N) {
    int i = blockIdx.x * blockDim.x + threadIdx.x;
    if (i < N) deg[i] = rsqrtf(deg[i]);    // deg >= 1 always (self-loops)
}

// ---------------------------------------------------------------------------
// Dense transform: Out[M,64] = A[M,64] @ W[64,64], f32 WMMA.
// One wave (32 lanes) computes a 16-row x 64-col strip of the output.
// M must be a multiple of 16 (N_NODES = 100000 = 6250 * 16).
// ---------------------------------------------------------------------------
__global__ __launch_bounds__(256) void k_gemm64(const float* __restrict__ A,
                                                const float* __restrict__ W,
                                                float* __restrict__ Out, int M) {
    const int lane  = threadIdx.x & 31;
    const int wave  = threadIdx.x >> 5;
    const int tileM = blockIdx.x * 8 + wave;           // 8 waves per 256-thread block
    if (tileM * 16 >= M) return;                       // uniform per-wave exit: EXEC stays all-1s

    const int rowBase = tileM * 16;
    const int l16     = lane & 15;
    const int hi      = lane >> 4;                     // 0 for lanes 0-15, 1 for 16-31
    const int row     = rowBase + l16;

    v8f acc[4];
    acc[0] = (v8f){}; acc[1] = (v8f){}; acc[2] = (v8f){}; acc[3] = (v8f){};

#if GCN_USE_F32_WMMA
    // A 16x4 f32 frag: lanes 0-15 hold K=k+0,k+1 ; lanes 16-31 hold K=k+2,k+3
    const int kOff = hi * 2;
    #pragma unroll
    for (int k = 0; k < 64; k += 4) {
        v2f a;
        a.x = A[row * 64 + k + kOff];
        a.y = A[row * 64 + k + kOff + 1];
        const float* Wr0 = W + (k + kOff) * 64;        // B rows K and K+1
        const float* Wr1 = W + (k + kOff + 1) * 64;
        #pragma unroll
        for (int j = 0; j < 4; ++j) {
            v2f b;
            b.x = Wr0[j * 16 + l16];
            b.y = Wr1[j * 16 + l16];
            acc[j] = __builtin_amdgcn_wmma_f32_16x16x4_f32(
                false, a, false, b, (short)0, acc[j], false, false);
        }
    }
#else
    // Fallback: f16 inputs, f32 accumulate (codegen-confirmed builtin).
    const int kb0 = hi * 8;                            // per-lane K base (A and B layouts)
    #pragma unroll
    for (int kb = 0; kb < 64; kb += 32) {
        v16h a;
        #pragma unroll
        for (int h = 0; h < 16; ++h) {
            int K = (h < 8) ? (kb0 + h) : (16 + kb0 + (h - 8));
            a[h] = (_Float16)A[row * 64 + kb + K];
        }
        #pragma unroll
        for (int j = 0; j < 4; ++j) {
            v16h b;
            #pragma unroll
            for (int h = 0; h < 16; ++h) {
                int K = (h < 8) ? (kb0 + h) : (16 + kb0 + (h - 8));
                b[h] = (_Float16)W[(kb + K) * 64 + j * 16 + l16];
            }
            acc[j] = __builtin_amdgcn_wmma_f32_16x16x32_f16(
                false, a, false, b, (short)0, acc[j], false, false);
        }
    }
#endif

    // C/D layout: VGPR r -> row (r) for lanes 0-15, row (r+8) for lanes 16-31; col = lane%16
    #pragma unroll
    for (int r = 0; r < 8; ++r) {
        int orow = rowBase + r + hi * 8;
        float* o = Out + (size_t)orow * 64 + l16;
        o[0]  = acc[0][r];
        o[16] = acc[1][r];
        o[32] = acc[2][r];
        o[48] = acc[3][r];
    }
}

// ---------------------------------------------------------------------------
// Self-loop init: agg[i,c] = t[i,c] * dinv[i]^2   (also zero-initializes agg)
// ---------------------------------------------------------------------------
__global__ __launch_bounds__(256) void k_selfloop(const float* __restrict__ t,
                                                  const float* __restrict__ dinv,
                                                  float* __restrict__ agg, int N) {
    int idx = blockIdx.x * blockDim.x + threadIdx.x;
    if (idx < N * 64) {
        int i = idx >> 6;
        float d = dinv[i];
        agg[idx] = t[idx] * d * d;
    }
}

// ---------------------------------------------------------------------------
// Edge scatter: 16 lanes per edge, float4 gather, scalar f32 atomics.
// agg[dst] += t[src] * dinv[src]*dinv[dst]
// ---------------------------------------------------------------------------
__global__ __launch_bounds__(256) void k_edge_scatter(const int* __restrict__ src,
                                                      const int* __restrict__ dst,
                                                      const float* __restrict__ dinv,
                                                      const float* __restrict__ t,
                                                      float* __restrict__ agg, int E) {
    long long tid = (long long)blockIdx.x * blockDim.x + threadIdx.x;
    int e = (int)(tid >> 4);
    if (e >= E) return;
    int sub = (int)(tid & 15);
    int s = src[e];
    int d = dst[e];
    float nrm = dinv[s] * dinv[d];
    const float4 v = *reinterpret_cast<const float4*>(t + (size_t)s * 64 + sub * 4);
    float* ap = agg + (size_t)d * 64 + sub * 4;
    atomicAdd(ap + 0, v.x * nrm);
    atomicAdd(ap + 1, v.y * nrm);
    atomicAdd(ap + 2, v.z * nrm);
    atomicAdd(ap + 3, v.w * nrm);
}

// ---------------------------------------------------------------------------
// Bias + ReLU
// ---------------------------------------------------------------------------
__global__ __launch_bounds__(256) void k_bias_relu(const float* __restrict__ agg,
                                                   const float* __restrict__ b,
                                                   float* __restrict__ h, int N) {
    int idx = blockIdx.x * blockDim.x + threadIdx.x;
    if (idx < N * 64) {
        int c = idx & 63;
        h[idx] = fmaxf(agg[idx] + b[c], 0.0f);
    }
}

// ---------------------------------------------------------------------------
// Final: out[i] = sigmoid( sum_c (agg[i,c] + b2[c]) * Wl[c] + bl )
// ---------------------------------------------------------------------------
__global__ __launch_bounds__(256) void k_final(const float* __restrict__ agg,
                                               const float* __restrict__ b2,
                                               const float* __restrict__ Wl,
                                               const float* __restrict__ bl,
                                               float* __restrict__ out, int N) {
    int i = blockIdx.x * blockDim.x + threadIdx.x;
    if (i >= N) return;
    float acc = bl[0];
    const float* h = agg + (size_t)i * 64;
    #pragma unroll
    for (int c = 0; c < 64; ++c) acc += (h[c] + b2[c]) * Wl[c];
    out[i] = 1.0f / (1.0f + __expf(-acc));
}

// ---------------------------------------------------------------------------
// Driver
// ---------------------------------------------------------------------------
extern "C" void kernel_launch(void* const* d_in, const int* in_sizes, int n_in,
                              void* d_out, int out_size, void* d_ws, size_t ws_size,
                              hipStream_t stream) {
    const float* x  = (const float*)d_in[0];
    const int*   ei = (const int*)  d_in[1];   // [2, E] flat: src then dst
    const float* W1 = (const float*)d_in[2];
    const float* b1 = (const float*)d_in[3];
    const float* W2 = (const float*)d_in[4];
    const float* b2 = (const float*)d_in[5];
    const float* Wl = (const float*)d_in[6];
    const float* bl = (const float*)d_in[7];
    float*       out = (float*)d_out;

    const int N = in_sizes[0] / 64;
    const int E = in_sizes[1] / 2;
    const int* src = ei;
    const int* dst = ei + E;

    float* ws   = (float*)d_ws;
    float* dinv = ws;                              // N floats (deg, then rsqrt in place)
    float* bufA = ws + N;                          // N*64 floats
    float* bufB = bufA + (size_t)N * 64;           // N*64 floats

    const int thr = 256;
    const int nBlk   = (N + thr - 1) / thr;
    const int eBlk   = (E + thr - 1) / thr;
    const int nfBlk  = (N * 64 + thr - 1) / thr;
    const int esBlk  = (int)(((long long)E * 16 + thr - 1) / thr);
    const int gemmBlk = (N / 16 + 7) / 8;          // 8 waves (M-tiles) per block

    // normalization
    k_init_deg<<<nBlk, thr, 0, stream>>>(dinv, N);
    k_deg_accum<<<eBlk, thr, 0, stream>>>(dst, dinv, E);
    k_rsqrt_inplace<<<nBlk, thr, 0, stream>>>(dinv, N);

    // layer 1: t = x@W1 ; agg = selfloop + edge-scatter ; h1 = relu(agg + b1)
    k_gemm64<<<gemmBlk, thr, 0, stream>>>(x, W1, bufA, N);
    k_selfloop<<<nfBlk, thr, 0, stream>>>(bufA, dinv, bufB, N);
    k_edge_scatter<<<esBlk, thr, 0, stream>>>(src, dst, dinv, bufA, bufB, E);
    k_bias_relu<<<nfBlk, thr, 0, stream>>>(bufB, b1, bufA, N);

    // layer 2: t = h1@W2 ; agg = selfloop + edge-scatter
    k_gemm64<<<gemmBlk, thr, 0, stream>>>(bufA, W2, bufB, N);
    k_selfloop<<<nfBlk, thr, 0, stream>>>(bufB, dinv, bufA, N);
    k_edge_scatter<<<esBlk, thr, 0, stream>>>(src, dst, dinv, bufB, bufA, E);

    // head: sigmoid((agg + b2) @ Wl + bl)
    k_final<<<nBlk, thr, 0, stream>>>(bufA, b2, Wl, bl, out, N);
}
